// ElementSpecificNN_274877907505
// MI455X (gfx1250) — matrix operand order repaired and checked by reading, not compiled, not run on hardware
//
#include <hip/hip_runtime.h>

typedef __attribute__((ext_vector_type(4)))  float    v4f;
typedef __attribute__((ext_vector_type(8)))  float    v8f;
typedef __attribute__((ext_vector_type(4)))  _Float16 v4h;
typedef __attribute__((ext_vector_type(8)))  _Float16 v8h;
typedef __attribute__((ext_vector_type(16))) _Float16 v16h;

#define BATCH   2048
#define NATOM   64
#define DIN     512
#define DH      512
#define NSPEC   4
#define TOTAL_ATOMS (BATCH * NATOM)   // 131072
#define TILE_M  128                   // atoms per block
#define MSUB    (TILE_M / 16)         // 8 M-sub-tiles (register M-extent per K-sweep)
#define NSUB    2                     // N-tiles per wave (16 waves x 2 x 16 = 512 cols)
#define NTHREADS 512
#define PITCHH  520                   // f16 pitch: 512 + 8 halfs (16B pad)
#define WSPITCH ((size_t)DH * DH)     // 262144 elements per (layer,species) matrix

#define LOG2E 1.4426950408889634f
#define LN2   0.6931471805599453f

__device__ __forceinline__ float softplus_f(float v) {
    // softplus(v) = max(v,0) + ln(1 + exp(-|v|)); log arg in [1,2] -> raw HW log safe
    float m = fmaxf(v, 0.0f);
    float e = __builtin_amdgcn_exp2f(-fabsf(v) * LOG2E);   // v_exp_f32
    float l = __builtin_amdgcn_logf(1.0f + e);             // v_log_f32 (log2)
    return fmaf(l, LN2, m);
}

__global__ void zero_counts_kernel(int* cnt) {
    if (threadIdx.x < NSPEC) cnt[threadIdx.x] = 0;
}

__global__ void gather_kernel(const int* __restrict__ charges,
                              const int* __restrict__ species,
                              int* __restrict__ cnt,
                              int* __restrict__ lists) {
    int i = blockIdx.x * blockDim.x + threadIdx.x;
    if (i >= TOTAL_ATOMS) return;
    int c = charges[i];
    int s = 0;
#pragma unroll
    for (int j = 1; j < NSPEC; ++j)
        if (c == species[j]) s = j;
    int pos = atomicAdd(&cnt[s], 1);
    lists[s * TOTAL_ATOMS + pos] = i;
}

// WT[s][n][k] = (f16) W[s][k][n]  -- coalesced writes, L2-resident strided reads
__global__ void transpose_kernel(const float* __restrict__ W,
                                 _Float16* __restrict__ WT) {
    size_t idx = (size_t)blockIdx.x * blockDim.x + threadIdx.x;  // over s,n,k
    int s = (int)(idx >> 18);
    int n = (int)((idx >> 9) & 511);
    int k = (int)(idx & 511);
    WT[idx] = (_Float16)W[(size_t)s * WSPITCH + (size_t)k * DH + n];
}

__global__ __launch_bounds__(NTHREADS)
void mlp_kernel(const float* __restrict__ x,
                const _Float16* __restrict__ wt1, const float* __restrict__ b1,
                const _Float16* __restrict__ wt2, const float* __restrict__ b2,
                const _Float16* __restrict__ wt3, const float* __restrict__ b3,
                const _Float16* __restrict__ wt4, const float* __restrict__ b4,
                const float* __restrict__ W5,    const float* __restrict__ b5,
                const int* __restrict__ cnt, const int* __restrict__ lists,
                float* __restrict__ energy) {
    __shared__ _Float16 bufA[TILE_M][PITCHH];   // 130 KB
    __shared__ _Float16 bufB[TILE_M][PITCHH];   // 130 KB
    __shared__ int      idxs[TILE_M];
    __shared__ float    partial[TILE_M][4];

    const int s     = blockIdx.y;
    const int tile0 = blockIdx.x * TILE_M;
    const int count = cnt[s];
    if (tile0 >= count) return;   // uniform per block: EXEC stays all-ones for WMMA

    const int t = threadIdx.x;
    if (t < TILE_M) {
        int slot = tile0 + t;
        idxs[t] = (slot < count) ? lists[s * TOTAL_ATOMS + slot] : -1;
    }
    __syncthreads();

    // Stage 128 input rows into LDS as f16 (coalesced b128 global loads).
    for (int q = t; q < TILE_M * (DIN / 4); q += NTHREADS) {
        int row = q >> 7;          // DIN/4 == 128
        int e4  = q & 127;
        int idx = idxs[row];
        int src = (idx < 0) ? 0 : idx;
        v4f val = *(const v4f*)(x + (size_t)src * DIN + e4 * 4);
        *(v4h*)(&bufA[row][e4 * 4]) = __builtin_convertvector(val, v4h);
    }
    __syncthreads();

    const int wave   = t >> 5;
    const int lane   = t & 31;
    const int mrow   = lane & 15;            // A row / B column within 16x16 tile
    const int khalf8 = (lane >> 4) << 3;     // unused (0 or 8); A also uses koff16
    const int koff16 = (lane >> 4) << 4;     // B: K base 0 or 16
    const int mbase  = (lane >> 4) << 3;     // C/D: +8 row offset for upper half
    const int col0   = wave * (NSUB * 16) + mrow;   // wave owns cols [wave*32, wave*32+32)

    auto layer = [&](const _Float16 (*cur)[PITCHH], _Float16 (*nxt)[PITCHH],
                     const _Float16* __restrict__ WT, const float* __restrict__ bias) {
        // 16 register-resident accumulator tiles: 2 N-tiles x 8 M-sub-tiles
        v8f acc[NSUB][MSUB];
#pragma unroll
        for (int j = 0; j < NSUB; ++j) {
            float bb = bias[col0 + j * 16];
#pragma unroll
            for (int mi = 0; mi < MSUB; ++mi)
                acc[j][mi] = (v8f){bb, bb, bb, bb, bb, bb, bb, bb};
        }

        // B pointer: WT[n][k] with n = col0 + j*16, 16 contiguous K halfs per lane
        const _Float16* wp = WT + (size_t)col0 * DH + koff16;

#pragma unroll 2
        for (int k = 0; k < DH; k += 32) {
            v16h bfrag[NSUB];
#pragma unroll
            for (int j = 0; j < NSUB; ++j)
                bfrag[j] = *(const v16h*)(wp + (size_t)j * 16 * DH + k);
#pragma unroll
            for (int mi = 0; mi < MSUB; ++mi) {
                // A fragment 16x32: per lane 16 contiguous K halfs (K +0..15 / +16..31)
                const _Float16* ap = &cur[mi * 16 + mrow][k + koff16];
                v8h lo = *(const v8h*)(ap);
                v8h hi = *(const v8h*)(ap + 8);
                v16h a = __builtin_shufflevector(lo, hi, 0, 1, 2, 3, 4, 5, 6, 7,
                                                 8, 9, 10, 11, 12, 13, 14, 15);
#pragma unroll
                for (int j = 0; j < NSUB; ++j)
                    acc[j][mi] = __builtin_amdgcn_wmma_f32_16x16x32_f16(
                        false, a, false, bfrag[j], (short)0, acc[j][mi], false, false);
            }
        }

#pragma unroll
        for (int mi = 0; mi < MSUB; ++mi)
#pragma unroll
            for (int j = 0; j < NSUB; ++j)
#pragma unroll
                for (int r = 0; r < 8; ++r)
                    nxt[mi * 16 + mbase + r][col0 + j * 16] =
                        (_Float16)softplus_f(acc[j][mi][r]);
        __syncthreads();
    };

    layer(bufA, bufB, wt1 + (size_t)s * WSPITCH, b1 + s * DH);
    layer(bufB, bufA, wt2 + (size_t)s * WSPITCH, b2 + s * DH);
    layer(bufA, bufB, wt3 + (size_t)s * WSPITCH, b3 + s * DH);
    layer(bufB, bufA, wt4 + (size_t)s * WSPITCH, b4 + s * DH);
    // final activations in bufA

    // Layer 5: [128 x 512] @ [512 x 1] + b5, deterministic tree reduction (f32).
    {
        int m    = t >> 2;    // 0..127 atom
        int part = t & 3;     // 4 chunks of 128
        const float* w5 = W5 + (size_t)s * DH;
        float sum = 0.0f;
#pragma unroll 8
        for (int k = part * 128; k < part * 128 + 128; ++k)
            sum += (float)bufA[m][k] * w5[k];
        partial[m][part] = sum;
    }
    __syncthreads();
    if (t < TILE_M) {
        float e = b5[s];
#pragma unroll
        for (int p = 0; p < 4; ++p) e += partial[t][p];
        int idx = idxs[t];
        if (idx >= 0) energy[idx] = e;
    }
}

__global__ void reduce_kernel(const float* __restrict__ energy,
                              float* __restrict__ out) {
    __shared__ float red[NATOM];
    int b = blockIdx.x;
    int n = threadIdx.x;
    red[n] = energy[(size_t)b * NATOM + n];
    __syncthreads();
    for (int sft = NATOM / 2; sft > 0; sft >>= 1) {
        if (n < sft) red[n] += red[n + sft];
        __syncthreads();
    }
    if (n == 0) out[b] = red[0];
}

extern "C" void kernel_launch(void* const* d_in, const int* in_sizes, int n_in,
                              void* d_out, int out_size, void* d_ws, size_t ws_size,
                              hipStream_t stream) {
    (void)in_sizes; (void)n_in; (void)out_size; (void)ws_size;
    const float* x       = (const float*)d_in[0];
    const int*   charges = (const int*)d_in[1];
    const int*   species = (const int*)d_in[2];
    const float* W1 = (const float*)d_in[3];  const float* b1 = (const float*)d_in[4];
    const float* W2 = (const float*)d_in[5];  const float* b2 = (const float*)d_in[6];
    const float* W3 = (const float*)d_in[7];  const float* b3 = (const float*)d_in[8];
    const float* W4 = (const float*)d_in[9];  const float* b4 = (const float*)d_in[10];
    const float* W5 = (const float*)d_in[11]; const float* b5 = (const float*)d_in[12];
    float* out = (float*)d_out;

    char* ws      = (char*)d_ws;
    int*  cnt     = (int*)ws;                                     // 4 ints (64B slot)
    int*  lists   = (int*)(ws + 64);                              // 4 * A ints = 2 MB
    float* energy = (float*)(ws + 64 + sizeof(int) * (size_t)NSPEC * TOTAL_ATOMS);
    _Float16* wt  = (_Float16*)(ws + 64 + sizeof(int) * (size_t)NSPEC * TOTAL_ATOMS
                                + sizeof(float) * (size_t)TOTAL_ATOMS);
    const size_t wlayer = (size_t)NSPEC * WSPITCH;                // halfs per layer (4 matrices)
    _Float16* wt1 = wt;
    _Float16* wt2 = wt + wlayer;
    _Float16* wt3 = wt + 2 * wlayer;
    _Float16* wt4 = wt + 3 * wlayer;

    zero_counts_kernel<<<1, 32, 0, stream>>>(cnt);
    gather_kernel<<<TOTAL_ATOMS / 256, 256, 0, stream>>>(charges, species, cnt, lists);

    const int tgrid = (int)(wlayer / 256);    // 4*512*512/256 = 4096 blocks
    transpose_kernel<<<tgrid, 256, 0, stream>>>(W1, wt1);
    transpose_kernel<<<tgrid, 256, 0, stream>>>(W2, wt2);
    transpose_kernel<<<tgrid, 256, 0, stream>>>(W3, wt3);
    transpose_kernel<<<tgrid, 256, 0, stream>>>(W4, wt4);

    dim3 grid(TOTAL_ATOMS / TILE_M, NSPEC);   // blocks past a species' count exit early
    mlp_kernel<<<grid, NTHREADS, 0, stream>>>(x, wt1, b1, wt2, b2, wt3, b3, wt4, b4,
                                              W5, b5, cnt, lists, energy);

    reduce_kernel<<<BATCH, NATOM, 0, stream>>>(energy, out);
}
